// MultiheadAttention_1726576854405
// MI455X (gfx1250) — matrix operand order repaired and checked by reading, not compiled, not run on hardware
//
#include <hip/hip_runtime.h>
#include <hip/hip_bf16.h>

// ---------------------------------------------------------------------------
// MI455X (gfx1250) T5-relative-bias multihead attention.
// f16 WMMA everywhere (v_wmma_f32_16x16x32_f16), flash-style attention
// (no [B,H,T,T] materialization), TDM async tile loads into LDS.
// ---------------------------------------------------------------------------

#define Tt 1024
#define Bb 4
#define Cc 1024
#define Hh 16
#define HD 64

typedef __attribute__((ext_vector_type(16))) _Float16 v16h;
typedef __attribute__((ext_vector_type(8)))  _Float16 h8;
typedef __attribute__((ext_vector_type(8)))  float    v8f;

typedef unsigned int u32x4 __attribute__((ext_vector_type(4)));
typedef int          i32x4 __attribute__((ext_vector_type(4)));
typedef int          i32x8 __attribute__((ext_vector_type(8)));

// ---------------------------------------------------------------------------
// WMMA helper: D = A(16x32 f16) * B(32x16 f16) + C(16x16 f32)
// ---------------------------------------------------------------------------
__device__ __forceinline__ v8f wmma_f16(v16h a, v16h b, v8f c) {
  return __builtin_amdgcn_wmma_f32_16x16x32_f16(false, a, false, b,
                                                (short)0, c, false, false);
}

// A-fragment (16x32): lane holds row (lane&15); halves 0..7 = K chunk at
// k0 + 8*(lane>>4), halves 8..15 = same chunk + 16.  Caller passes p already
// offset to row*ld + k0 + ((lane>>4)<<3).
__device__ __forceinline__ v16h frag_a(const _Float16* p) {
  h8 lo = *(const h8*)(p);
  h8 hi = *(const h8*)(p + 16);
  v16h f;
#pragma unroll
  for (int i = 0; i < 8; ++i) { f[i] = lo[i]; f[i + 8] = hi[i]; }
  return f;
}

// B-fragment (32x16): lane holds column (lane&15); halves 0..7 = K chunk at
// k0 + 8*(lane>>4), halves 8..15 = same chunk + 16.  Caller passes p at row*ld + k0 + ((lane>>4)<<3).
__device__ __forceinline__ v16h frag_b(const _Float16* p) {
  h8 lo = *(const h8*)(p);
  h8 hi = *(const h8*)(p + 16);
  v16h f;
#pragma unroll
  for (int i = 0; i < 8; ++i) { f[i] = lo[i]; f[i + 8] = hi[i]; }
  return f;
}

// ---------------------------------------------------------------------------
// Tensor Data Mover: DMA a [tile_d1 x tile_d0] f16 tile (row stride src_stride
// elems) from global into LDS, padding each row to lds_pitch elems.
// Falls back to a plain copy by the calling wave if the builtin is absent.
// ---------------------------------------------------------------------------
#if __has_builtin(__builtin_amdgcn_tensor_load_to_lds)
#define HAVE_TDM 1
#else
#define HAVE_TDM 0
#endif

__device__ __forceinline__ void tile_to_lds(_Float16* lds, const _Float16* g,
                                            unsigned tile_d0, unsigned tile_d1,
                                            unsigned src_stride, unsigned lds_pitch,
                                            unsigned pad_interval_code, int lane) {
#if HAVE_TDM
  (void)lane;
  unsigned pad_dw = ((lds_pitch - tile_d0) * 2u) >> 2;   // padding in DWORDs
  unsigned long long ga = (unsigned long long)(const void*)g;
  u32x4 g0;
  g0[0] = 1u;                                            // count=1, user mode
  g0[1] = (unsigned)(unsigned long long)(void*)lds;      // LDS byte offset
  g0[2] = (unsigned)(ga & 0xFFFFFFFFull);                // global_addr[31:0]
  g0[3] = (unsigned)((ga >> 32) & 0x01FFFFFFull) | 0x80000000u; // addr[56:32] | type=2
  i32x8 g1;
  unsigned w0 = (1u << 16);                              // data_size = 2 bytes
  if (pad_dw)
    w0 |= (1u << 20) | (pad_interval_code << 22) | ((pad_dw - 1u) << 25);
  g1[0] = (int)w0;                                       // wg_mask=0 (no cluster)
  g1[1] = (int)((tile_d0 & 0xFFFFu) << 16);              // tensor_dim0 lo
  g1[2] = (int)((tile_d0 >> 16) | ((tile_d1 & 0xFFFFu) << 16)); // tensor_dim0 hi | dim1 lo
  g1[3] = (int)((tile_d1 >> 16) | ((tile_d0 & 0xFFFFu) << 16)); // dim1 hi | tile_dim0
  g1[4] = (int)(tile_d1 & 0xFFFFu);                      // tile_dim1 (tile_dim2=0)
  g1[5] = (int)src_stride;                               // tensor_dim0_stride lo
  g1[6] = 0;
  g1[7] = 0;
  i32x4 z4 = {0, 0, 0, 0};
#if __clang_major__ >= 23
  i32x8 z8 = {0, 0, 0, 0, 0, 0, 0, 0};
  __builtin_amdgcn_tensor_load_to_lds(g0, g1, z4, z4, z8, 0);
#else
  __builtin_amdgcn_tensor_load_to_lds(g0, g1, z4, z4, 0);
#endif
#else
  (void)pad_interval_code;
  for (unsigned r = (unsigned)lane; r < tile_d1; r += 32)
    for (unsigned c = 0; c < tile_d0; ++c)
      lds[r * lds_pitch + c] = g[(size_t)r * src_stride + c];
#endif
}

__device__ __forceinline__ void wait_tdm() {
#if HAVE_TDM && __has_builtin(__builtin_amdgcn_s_wait_tensorcnt)
  __builtin_amdgcn_s_wait_tensorcnt(0);
#endif
}

// ---------------------------------------------------------------------------
// f32 -> f16 conversion (grid-stride)
// ---------------------------------------------------------------------------
__global__ void cvt_f16_kernel(const float* __restrict__ s,
                               _Float16* __restrict__ d, int n) {
  for (int i = blockIdx.x * blockDim.x + threadIdx.x; i < n;
       i += gridDim.x * blockDim.x)
    d[i] = (_Float16)s[i];
}

// ---------------------------------------------------------------------------
// QKV projection: Y_p = X @ W_p^T + b_p   (p = 0,1,2 = q,k,v)
// X: [4096,1024] f16, W: [1024,1024] f16 row-major (row = out channel, col = k)
// Output: head layout [B,H,T,HD] f16; q additionally scaled by HD^-0.5.
// One workgroup = 64x64 output tile, 4 waves of 16 rows each.
// ---------------------------------------------------------------------------
__global__ __launch_bounds__(128) void qkv_gemm_kernel(
    const _Float16* __restrict__ x16, const _Float16* __restrict__ w16,
    const float* __restrict__ qb, const float* __restrict__ kb,
    const float* __restrict__ vb, _Float16* __restrict__ q16,
    _Float16* __restrict__ k16o, _Float16* __restrict__ v16o) {
  __shared__ __align__(16) _Float16 wt[64][40];  // 32 data + 8 pad per row
  const int p = blockIdx.z;
  const int n0 = blockIdx.x * 64, m0 = blockIdx.y * 64;
  const int wave = threadIdx.x >> 5, lane = threadIdx.x & 31;
  const int lo = lane & 15, hl = lane >> 4;
  const _Float16* wp = w16 + (size_t)p * Cc * Cc;
  v8f acc[4];
#pragma unroll
  for (int nt = 0; nt < 4; ++nt)
#pragma unroll
    for (int r = 0; r < 8; ++r) acc[nt][r] = 0.f;
  const int m = m0 + wave * 16 + lo;
  const _Float16* xrow = x16 + (size_t)m * Cc;

  for (int k0 = 0; k0 < Cc; k0 += 32) {
    __syncthreads();                       // protect LDS reuse
    if (wave == 0)
      tile_to_lds(&wt[0][0], wp + (size_t)n0 * Cc + k0, 32, 64, Cc, 40, 3, lane);
    v16h af = frag_a(xrow + k0 + (hl << 3));
    if (k0 + 32 < Cc) __builtin_prefetch(xrow + k0 + 32, 0, 1);
    wait_tdm();
    __syncthreads();
    // preload all B fragments, then issue WMMAs back-to-back (pipelined)
    v16h bf[4];
#pragma unroll
    for (int nt = 0; nt < 4; ++nt) bf[nt] = frag_b(&wt[nt * 16 + lo][hl << 3]);
#pragma unroll
    for (int nt = 0; nt < 4; ++nt) acc[nt] = wmma_f16(af, bf[nt], acc[nt]);
  }

  const float scale = (p == 0) ? 0.125f : 1.0f;  // HD^-0.5 = 1/8
  const float* bias = (p == 0) ? qb : (p == 1) ? kb : vb;
  _Float16* dst = (p == 0) ? q16 : (p == 1) ? k16o : v16o;
#pragma unroll
  for (int nt = 0; nt < 4; ++nt)
#pragma unroll
    for (int r = 0; r < 8; ++r) {
      int n = n0 + nt * 16 + lo;
      int mm = m0 + wave * 16 + r + 8 * hl;
      float v = (acc[nt][r] + bias[n]) * scale;
      int t = mm >> 2, bi = mm & 3;          // m = t*B + b, B=4
      int hh = n >> 6, d = n & 63;           // n = h*64 + d
      dst[(((size_t)bi * Hh + hh) * Tt + t) * HD + d] = (_Float16)v;
    }
}

// ---------------------------------------------------------------------------
// Relative-position bias table: bias_tab[h][delta + 1023], delta = k - q.
// Precomputes the log-bucketed T5 bias once (kills per-element transcendentals).
// ---------------------------------------------------------------------------
__global__ void bias_kernel(const float* __restrict__ rel_bias,
                            float* __restrict__ bias_tab) {
  const int h = blockIdx.x;
  for (int i = threadIdx.x; i < 2 * Tt - 1; i += blockDim.x) {
    int rel = i - (Tt - 1);
    int base = (rel > 0) ? 16 : 0;           // NUM_BUCKETS/2
    int a = rel < 0 ? -rel : rel;
    int off;
    if (a < 8) {
      off = a;                               // max_exact = 8
    } else {
      float rl = 8.f + __logf((float)a * 0.125f) * (8.f / __logf(16.f));
      off = (int)rl;
      if (off > 15) off = 15;
    }
    bias_tab[h * (2 * Tt - 1) + i] = rel_bias[(base + off) * Hh + h];
  }
}

// ---------------------------------------------------------------------------
// Gate precompute: gate_a_1[b,h,t] = ga*(gb*grep_a[h]-1)+2 from qh @ grep_w^T.
// ---------------------------------------------------------------------------
__global__ void gate_kernel(const _Float16* __restrict__ q16,
                            const float* __restrict__ grep_w,
                            const float* __restrict__ grep_b,
                            const float* __restrict__ grep_a,
                            float* __restrict__ gate_tab) {
  int idx = blockIdx.x * blockDim.x + threadIdx.x;  // (b*H + h)*T + t
  if (idx >= Bb * Hh * Tt) return;
  int h = (idx >> 10) & (Hh - 1);
  const _Float16* qv = q16 + (size_t)idx * HD;
  float g[8];
#pragma unroll
  for (int j = 0; j < 8; ++j) g[j] = grep_b[j];
  for (int d = 0; d < HD; ++d) {
    float qd = (float)qv[d];
#pragma unroll
    for (int j = 0; j < 8; ++j) g[j] += qd * grep_w[j * HD + d];
  }
  float s0 = g[0] + g[1] + g[2] + g[3];
  float s1 = g[4] + g[5] + g[6] + g[7];
  float ga = 1.f / (1.f + __expf(-s0));
  float gb = 1.f / (1.f + __expf(-s1));
  gate_tab[idx] = ga * (gb * grep_a[h] - 1.f) + 2.f;
}

// ---------------------------------------------------------------------------
// Flash attention: one workgroup per (b*H+h, 64-row q block); 4 waves of
// 16 q rows each. Streams 64-key blocks: S = Q K^T (WMMA), + gate*bias,
// online softmax (shfl_xor row reductions), O += P V (WMMA via LDS restage).
// ---------------------------------------------------------------------------
__global__ __launch_bounds__(128) void flash_attn_kernel(
    const _Float16* __restrict__ q16, const _Float16* __restrict__ k16,
    const _Float16* __restrict__ v16, const float* __restrict__ gate_tab,
    const float* __restrict__ bias_tab, _Float16* __restrict__ att16) {
  __shared__ __align__(16) _Float16 kt[64][72];      // K tile, padded pitch
  __shared__ __align__(16) _Float16 vt[64][72];      // V tile transposed [d][j]
  __shared__ __align__(16) _Float16 pt[4][16][72];   // per-wave P restage
  __shared__ float sbias[2 * Tt - 1];

  const int bh = blockIdx.y;                 // b*H + h
  const int b = bh >> 4, h = bh & 15;
  const int q0 = blockIdx.x * 64;
  const int wave = threadIdx.x >> 5, lane = threadIdx.x & 31;
  const int lo = lane & 15, hl = lane >> 4;

  for (int i = threadIdx.x; i < 2 * Tt - 1; i += 128)
    sbias[i] = bias_tab[h * (2 * Tt - 1) + i];

  const _Float16* qbase = q16 + (size_t)bh * Tt * HD;
  const _Float16* kbase = k16 + (size_t)bh * Tt * HD;
  const _Float16* vbase = v16 + (size_t)bh * Tt * HD;

  // Q fragments for this wave's 16 rows (q already scaled)
  const int qrow = q0 + wave * 16 + lo;
  v16h qf0 = frag_a(qbase + (size_t)qrow * HD + 0 + (hl << 3));
  v16h qf1 = frag_a(qbase + (size_t)qrow * HD + 32 + (hl << 3));

  float gate[8];
#pragma unroll
  for (int r = 0; r < 8; ++r)
    gate[r] = gate_tab[(size_t)bh * Tt + q0 + wave * 16 + r + 8 * hl];

  float m_run[8], l_run[8];
  v8f o[4];
#pragma unroll
  for (int r = 0; r < 8; ++r) { m_run[r] = -1e30f; l_run[r] = 0.f; }
#pragma unroll
  for (int nt = 0; nt < 4; ++nt)
#pragma unroll
    for (int r = 0; r < 8; ++r) o[nt][r] = 0.f;

  for (int kb = 0; kb < Tt; kb += 64) {
    __syncthreads();                         // previous tile fully consumed
    if (wave == 0)                           // rows contiguous: stride = HD
      tile_to_lds(&kt[0][0], kbase + (size_t)kb * HD, 64, 64, 64, 72, 4, lane);
    if (kb + 64 < Tt) {
      __builtin_prefetch(kbase + (size_t)(kb + 64) * HD, 0, 1);
      __builtin_prefetch(vbase + (size_t)(kb + 64) * HD, 0, 1);
    }
    {  // cooperative transposed V load: vt[d][j] = V[kb+j][d]
      int j = threadIdx.x >> 1;
      int d0 = (threadIdx.x & 1) * 32;
      const h8* vp8 = (const h8*)(vbase + (size_t)(kb + j) * HD + d0);
#pragma unroll
      for (int c = 0; c < 4; ++c) {
        h8 vv = vp8[c];
#pragma unroll
        for (int i = 0; i < 8; ++i) vt[d0 + c * 8 + i][j] = vv[i];
      }
    }
    wait_tdm();
    __syncthreads();

    // S = Q @ K^T : 16x64 per wave (contraction over HD=64 -> 2 chained WMMA)
    // Preload all 8 K fragments, then issue the 8 WMMAs back-to-back.
    v16h kf[8];
#pragma unroll
    for (int nt = 0; nt < 4; ++nt) {
      kf[2 * nt + 0] = frag_b(&kt[nt * 16 + lo][0 + (hl << 3)]);
      kf[2 * nt + 1] = frag_b(&kt[nt * 16 + lo][32 + (hl << 3)]);
    }
    v8f s[4];
#pragma unroll
    for (int nt = 0; nt < 4; ++nt) {
      v8f z;
#pragma unroll
      for (int r = 0; r < 8; ++r) z[r] = 0.f;
      z = wmma_f16(qf0, kf[2 * nt + 0], z);
      z = wmma_f16(qf1, kf[2 * nt + 1], z);
      s[nt] = z;
    }

    // + gate_a_1(q) * rel_bias(k - q)
#pragma unroll
    for (int nt = 0; nt < 4; ++nt)
#pragma unroll
      for (int r = 0; r < 8; ++r) {
        int tq = q0 + wave * 16 + r + 8 * hl;
        int tk = kb + nt * 16 + lo;
        s[nt][r] += gate[r] * sbias[tk - tq + (Tt - 1)];
      }

    // online softmax (rows live across 16-lane halves; reduce with shfl_xor)
    float mx[8];
#pragma unroll
    for (int r = 0; r < 8; ++r) {
      mx[r] = fmaxf(fmaxf(s[0][r], s[1][r]), fmaxf(s[2][r], s[3][r]));
#pragma unroll
      for (int msk = 1; msk < 16; msk <<= 1)
        mx[r] = fmaxf(mx[r], __shfl_xor(mx[r], msk, 32));
    }
    float alpha[8], rs[8];
#pragma unroll
    for (int r = 0; r < 8; ++r) {
      float mn = fmaxf(m_run[r], mx[r]);
      alpha[r] = __expf(m_run[r] - mn);
      m_run[r] = mn;
      rs[r] = 0.f;
    }
#pragma unroll
    for (int nt = 0; nt < 4; ++nt)
#pragma unroll
      for (int r = 0; r < 8; ++r) {
        float pv = __expf(s[nt][r] - m_run[r]);
        rs[r] += pv;
        pt[wave][r + 8 * hl][nt * 16 + lo] = (_Float16)pv;
      }
#pragma unroll
    for (int r = 0; r < 8; ++r) {
#pragma unroll
      for (int msk = 1; msk < 16; msk <<= 1)
        rs[r] += __shfl_xor(rs[r], msk, 32);
      l_run[r] = l_run[r] * alpha[r] + rs[r];
    }
#pragma unroll
    for (int nt = 0; nt < 4; ++nt)
#pragma unroll
      for (int r = 0; r < 8; ++r) o[nt][r] *= alpha[r];

    // O += P @ V  (contraction over 64 keys; P restaged via LDS, same wave)
    v16h pf0 = frag_a(&pt[wave][lo][0 + (hl << 3)]);
    v16h pf1 = frag_a(&pt[wave][lo][32 + (hl << 3)]);
    v16h vf[8];
#pragma unroll
    for (int nt = 0; nt < 4; ++nt) {
      vf[2 * nt + 0] = frag_b(&vt[nt * 16 + lo][0 + (hl << 3)]);
      vf[2 * nt + 1] = frag_b(&vt[nt * 16 + lo][32 + (hl << 3)]);
    }
#pragma unroll
    for (int nt = 0; nt < 4; ++nt) {
      o[nt] = wmma_f16(pf0, vf[2 * nt + 0], o[nt]);
      o[nt] = wmma_f16(pf1, vf[2 * nt + 1], o[nt]);
    }
  }

  // finalize: divide by row sums, store to [T,B,C] f16 for output projection
#pragma unroll
  for (int r = 0; r < 8; ++r) l_run[r] = 1.0f / l_run[r];
#pragma unroll
  for (int nt = 0; nt < 4; ++nt)
#pragma unroll
    for (int r = 0; r < 8; ++r) {
      int t = q0 + wave * 16 + r + 8 * hl;
      int n = h * HD + nt * 16 + lo;
      att16[((size_t)t * Bb + b) * Cc + n] = (_Float16)(o[nt][r] * l_run[r]);
    }
}

// ---------------------------------------------------------------------------
// Output projection: out = A @ Wo^T + out_b  (f32 result straight to d_out)
// ---------------------------------------------------------------------------
__global__ __launch_bounds__(128) void out_gemm_kernel(
    const _Float16* __restrict__ a16, const _Float16* __restrict__ w16o,
    const float* __restrict__ ob, float* __restrict__ out) {
  __shared__ __align__(16) _Float16 wt[64][40];
  const int n0 = blockIdx.x * 64, m0 = blockIdx.y * 64;
  const int wave = threadIdx.x >> 5, lane = threadIdx.x & 31;
  const int lo = lane & 15, hl = lane >> 4;
  v8f acc[4];
#pragma unroll
  for (int nt = 0; nt < 4; ++nt)
#pragma unroll
    for (int r = 0; r < 8; ++r) acc[nt][r] = 0.f;
  const int m = m0 + wave * 16 + lo;
  const _Float16* arow = a16 + (size_t)m * Cc;

  for (int k0 = 0; k0 < Cc; k0 += 32) {
    __syncthreads();
    if (wave == 0)
      tile_to_lds(&wt[0][0], w16o + (size_t)n0 * Cc + k0, 32, 64, Cc, 40, 3, lane);
    v16h af = frag_a(arow + k0 + (hl << 3));
    if (k0 + 32 < Cc) __builtin_prefetch(arow + k0 + 32, 0, 1);
    wait_tdm();
    __syncthreads();
    v16h bf[4];
#pragma unroll
    for (int nt = 0; nt < 4; ++nt) bf[nt] = frag_b(&wt[nt * 16 + lo][hl << 3]);
#pragma unroll
    for (int nt = 0; nt < 4; ++nt) acc[nt] = wmma_f16(af, bf[nt], acc[nt]);
  }
#pragma unroll
  for (int nt = 0; nt < 4; ++nt)
#pragma unroll
    for (int r = 0; r < 8; ++r) {
      int n = n0 + nt * 16 + lo;
      int mm = m0 + wave * 16 + r + 8 * hl;
      out[(size_t)mm * Cc + n] = acc[nt][r] + ob[n];
    }
}

// ---------------------------------------------------------------------------
// Workspace layout (bytes)
// ---------------------------------------------------------------------------
static constexpr size_t OFF_X16  = 0;                        // query f16 [4096,1024]
static constexpr size_t OFF_W16  = OFF_X16 + 8388608;        // 4 x [1024,1024] f16
static constexpr size_t OFF_Q16  = OFF_W16 + 8388608;        // [B,H,T,HD] f16
static constexpr size_t OFF_K16  = OFF_Q16 + 8388608;
static constexpr size_t OFF_V16  = OFF_K16 + 8388608;
static constexpr size_t OFF_A16  = OFF_V16 + 8388608;        // attn out [T,B,C] f16
static constexpr size_t OFF_GATE = OFF_A16 + 8388608;        // [B,H,T] f32
static constexpr size_t OFF_BIAS = OFF_GATE + 262144;        // [H, 2T-1] f32

extern "C" void kernel_launch(void* const* d_in, const int* in_sizes, int n_in,
                              void* d_out, int out_size, void* d_ws,
                              size_t ws_size, hipStream_t stream) {
  (void)in_sizes; (void)n_in; (void)out_size; (void)ws_size;
  const float* query    = (const float*)d_in[0];
  const float* q_w      = (const float*)d_in[1];
  const float* q_b      = (const float*)d_in[2];
  const float* k_w      = (const float*)d_in[3];
  const float* k_b      = (const float*)d_in[4];
  const float* v_w      = (const float*)d_in[5];
  const float* v_b      = (const float*)d_in[6];
  const float* out_w    = (const float*)d_in[7];
  const float* out_b    = (const float*)d_in[8];
  const float* rel_bias = (const float*)d_in[9];
  const float* grep_w   = (const float*)d_in[10];
  const float* grep_b   = (const float*)d_in[11];
  const float* grep_a   = (const float*)d_in[12];

  char* ws = (char*)d_ws;
  _Float16* x16  = (_Float16*)(ws + OFF_X16);
  _Float16* w16  = (_Float16*)(ws + OFF_W16);   // [q,k,v,o]
  _Float16* q16  = (_Float16*)(ws + OFF_Q16);
  _Float16* k16  = (_Float16*)(ws + OFF_K16);
  _Float16* v16  = (_Float16*)(ws + OFF_V16);
  _Float16* a16  = (_Float16*)(ws + OFF_A16);
  float* gate_tab = (float*)(ws + OFF_GATE);
  float* bias_tab = (float*)(ws + OFF_BIAS);

  // 1) f32 -> f16 staging
  cvt_f16_kernel<<<2048, 256, 0, stream>>>(query, x16, Tt * Bb * Cc);
  cvt_f16_kernel<<<1024, 256, 0, stream>>>(q_w,   w16 + 0 * Cc * Cc, Cc * Cc);
  cvt_f16_kernel<<<1024, 256, 0, stream>>>(k_w,   w16 + 1 * (size_t)Cc * Cc, Cc * Cc);
  cvt_f16_kernel<<<1024, 256, 0, stream>>>(v_w,   w16 + 2 * (size_t)Cc * Cc, Cc * Cc);
  cvt_f16_kernel<<<1024, 256, 0, stream>>>(out_w, w16 + 3 * (size_t)Cc * Cc, Cc * Cc);

  // 2) QKV projections (WMMA, TDM-staged weight tiles)
  qkv_gemm_kernel<<<dim3(Cc / 64, (Tt * Bb) / 64, 3), 128, 0, stream>>>(
      x16, w16, q_b, k_b, v_b, q16, k16, v16);

  // 3) bias table + gating scalars
  bias_kernel<<<Hh, 256, 0, stream>>>(rel_bias, bias_tab);
  gate_kernel<<<(Bb * Hh * Tt) / 256, 256, 0, stream>>>(q16, grep_w, grep_b,
                                                        grep_a, gate_tab);

  // 4) flash attention (WMMA + online softmax)
  flash_attn_kernel<<<dim3(Tt / 64, Bb * Hh), 128, 0, stream>>>(
      q16, k16, v16, gate_tab, bias_tab, a16);

  // 5) output projection
  out_gemm_kernel<<<dim3(Cc / 64, (Tt * Bb) / 64), 128, 0, stream>>>(
      a16, w16 + 3 * (size_t)Cc * Cc, out_b, (float*)d_out);
}
